// SUSTeR_28020366639711
// MI455X (gfx1250) — matrix operand order repaired and checked by reading, not compiled, not run on hardware
//
#include <hip/hip_runtime.h>

// SUSTeR forward for MI455X (gfx1250, wave32, WMMA).
// One block per batch; sequential T-scan kept LDS-resident; both per-step
// GEMMs (logits = proj@prevT, delta = densT@proj) on v_wmma_f32_16x16x32_f16.
// Fragments move as 16B v8h chunks (ISA A-fragment K-map is contiguous in
// 8-half runs); density is COLUMN-major [p][s] so D-fragment rows pack into
// ds_store_b128; the per-step delta buffer is TRANSPOSED [e][p] so the
// cross-wave reduction RMWs are float4-vectorized (D-fragment v=0..7 hit
// consecutive p).

typedef _Float16 f16;
typedef __attribute__((ext_vector_type(16))) _Float16 v16h;
typedef __attribute__((ext_vector_type(8)))  _Float16 v8h;
typedef __attribute__((ext_vector_type(8)))  float    v8f;
typedef __attribute__((ext_vector_type(4)))  float    v4f;

union Frag16 { v16h f; v8h h2[2]; };

#define NB   16
#define NT   12
#define NS   1024
#define NC   3
#define NP   64
#define NE   32
#define NTGT 256
#define THREADS 256
#define NWAVES  8
#define CHUNK   32                       // s-rows per chunk = K of GEMM2
#define CPW     (NS / (NWAVES * CHUNK))  // 4 chunks per wave

__global__ __launch_bounds__(THREADS)
void suster_fwd(const float* __restrict__ obs,        // (B,T,S,C)
                const float* __restrict__ mean_state, // (P,E)
                const float* __restrict__ W_obs,      // (C,E)
                const float* __restrict__ b_obs,      // (E)
                const float* __restrict__ W_out,      // (P*E, TGT)
                const float* __restrict__ b_out,      // (TGT)
                float* __restrict__ out)              // out(4096) ++ base(4096)
{
  __shared__ f16   s_prev[NP * NE];                // carried state [p][e] (4 KB)
  __shared__ float s_delta[NE * NP];               // step reduce, [e][p]! (8 KB)
  __shared__ f16   s_proj[NWAVES][CHUNK * NE];     // proj row-major [s][e] (16 KB)
  __shared__ f16   s_dens[NWAVES][NP * CHUNK];     // density col-major [p][s] (32 KB)
  __shared__ float s_wobs[NC * NE];
  __shared__ float s_bobs[NE];

  const int tid  = threadIdx.x;
  const int lane = tid & 31;
  const int wave = tid >> 5;
  const int b    = blockIdx.x;
  const int ln16 = lane & 15;
  const int hi8  = (lane >> 4) << 3;   // 0 or 8  (C/D row offset, A K-offset)
  const int hi16 = (lane >> 4) << 4;   // 0 or 16 (unused; B K-offset is hi8 too)

  // full_state accumulator distributed across threads: 8 (p,e) elems each,
  // flat order i = p*NE + e.
  float fullacc[8];
  #pragma unroll
  for (int j = 0; j < 8; ++j) {
    int i = tid * 8 + j;
    float m = mean_state[i];
    s_prev[i] = (f16)m;
    fullacc[j] = m;           // full_state = ms + sum_t w_t * delta_t
  }
  if (tid < NC * NE) s_wobs[tid] = W_obs[tid];
  if (tid < NE)      s_bobs[tid] = b_obs[tid];
  // W_out column only needed after the scan: warm it early (global_prefetch).
  __builtin_prefetch(&W_out[tid], 0, 1);
  __syncthreads();

  #pragma unroll 1
  for (int t = 0; t < NT; ++t) {
    // GEMM1 B fragments (prevT, K=e x N=p): constant for the whole step t.
    // B layout: half h -> (K = hi8+h / 16+hi8+h-8, N = ln16); two 8-half runs.
    Frag16 bPrev[4];
    #pragma unroll
    for (int pt = 0; pt < 4; ++pt) {
      const int base = (pt * 16 + ln16) * NE + hi8;
      bPrev[pt].h2[0] = *(const v8h*)&s_prev[base];
      bPrev[pt].h2[1] = *(const v8h*)&s_prev[base + 16];
    }

    // per-wave delta partials: 4 P-tiles x 2 E-tiles of 16x16 f32
    v8f dAcc[4][2];
    #pragma unroll
    for (int pt = 0; pt < 4; ++pt)
      #pragma unroll
      for (int et = 0; et < 2; ++et)
        #pragma unroll
        for (int v = 0; v < 8; ++v) dAcc[pt][et][v] = 0.0f;

    v8f cz;
    #pragma unroll
    for (int v = 0; v < 8; ++v) cz[v] = 0.0f;

    #pragma unroll 1
    for (int c = 0; c < CPW; ++c) {
      const int s0 = (wave * CPW + c) * CHUNK;

      // ---- proj = obs @ W_obs + b_obs : one s-row per lane, f16 into LDS ----
      unsigned long long mball;  // bit r = row r of this chunk is non-zero
      {
        const size_t gbase = (((size_t)b * NT + t) * NS + (s0 + lane)) * NC;
        const float o0 = obs[gbase + 0];
        const float o1 = obs[gbase + 1];
        const float o2 = obs[gbase + 2];
        mball = __ballot(o0 != 0.0f || o1 != 0.0f || o2 != 0.0f);
        v8h row[4];
        #pragma unroll
        for (int e = 0; e < NE; ++e) {
          float p = s_bobs[e] + o0 * s_wobs[e] + o1 * s_wobs[NE + e] +
                    o2 * s_wobs[2 * NE + e];
          row[e >> 3][e & 7] = (f16)p;
        }
        #pragma unroll
        for (int q = 0; q < 4; ++q)
          *(v8h*)&s_proj[wave][lane * NE + q * 8] = row[q];
      }

      // ---- GEMM1 (logits = proj @ prevT) + in-register softmax, per s-tile --
      #pragma unroll
      for (int st = 0; st < 2; ++st) {
        // A: proj tile 16x32 (M=s, K=e). ISA K-map = {hi8+h, 16+hi8+h}:
        // two contiguous 16B chunks per lane.
        Frag16 aF;
        {
          const int base = (st * 16 + ln16) * NE + hi8;
          aF.h2[0] = *(const v8h*)&s_proj[wave][base];
          aF.h2[1] = *(const v8h*)&s_proj[wave][base + 16];
        }
        v8f logit[4];
        #pragma unroll
        for (int pt = 0; pt < 4; ++pt)
          logit[pt] = __builtin_amdgcn_wmma_f32_16x16x32_f16(
              false, aF.f, false, bPrev[pt].f, (short)0, cz, false, false);

        // softmax over P=64: per VGPR v, row M = v + hi8; columns live across
        // the 4 pt-tiles and 16 lanes of this half -> shfl_xor butterflies.
        // Lane holds rows v=0..7 of column p = pt*16+ln16, so the densities
        // pack into one contiguous v8h per pt in the [p][s] layout.
        v8h packD[4];
        #pragma unroll
        for (int v = 0; v < 8; ++v) {
          float mx = fmaxf(fmaxf(logit[0][v], logit[1][v]),
                           fmaxf(logit[2][v], logit[3][v]));
          #pragma unroll
          for (int off = 1; off < 16; off <<= 1)
            mx = fmaxf(mx, __shfl_xor(mx, off, 32));
          float ex[4];
          float sm = 0.0f;
          #pragma unroll
          for (int pt = 0; pt < 4; ++pt) {
            ex[pt] = __expf(logit[pt][v] - mx);
            sm += ex[pt];
          }
          #pragma unroll
          for (int off = 1; off < 16; off <<= 1)
            sm += __shfl_xor(sm, off, 32);
          const int srow = st * 16 + v + hi8;       // s-row within chunk
          const float mk =
              ((mball >> srow) & 1ull) ? (1.0f / sm) : 0.0f;  // mask folded in
          #pragma unroll
          for (int pt = 0; pt < 4; ++pt)
            packD[pt][v] = (f16)(ex[pt] * mk);
        }
        #pragma unroll
        for (int pt = 0; pt < 4; ++pt)
          *(v8h*)&s_dens[wave][(pt * 16 + ln16) * CHUNK + st * 16 + hi8] =
              packD[pt];
      }

      // ---- GEMM2: delta += densT (P x 32) @ proj (32 x E) ----
      // B fragments (proj as K=s x N=e) depend only on et: gather once per
      // chunk (strided transpose gather -- the only element-wise LDS left).
      Frag16 bProj[2];
      #pragma unroll
      for (int et = 0; et < 2; ++et)
        #pragma unroll
        for (int h = 0; h < 16; ++h)
          bProj[et].f[h] = s_proj[wave][(hi8 + (h & 7) + ((h & 8) << 1)) * NE + et * 16 + ln16];

      #pragma unroll
      for (int pt = 0; pt < 4; ++pt) {
        // A: densT tile 16x32 (M=p, K=s) from col-major dens[p][s]:
        // contiguous 16B chunks again.
        Frag16 aF;
        {
          const int base = (pt * 16 + ln16) * CHUNK + hi8;
          aF.h2[0] = *(const v8h*)&s_dens[wave][base];
          aF.h2[1] = *(const v8h*)&s_dens[wave][base + 16];
        }
        #pragma unroll
        for (int et = 0; et < 2; ++et)
          dAcc[pt][et] = __builtin_amdgcn_wmma_f32_16x16x32_f16(
              false, aF.f, false, bProj[et].f, (short)0, dAcc[pt][et], false,
              false);
      }
    }  // chunks

    // ---- cross-wave delta reduction into s_delta[e][p] -------------------
    // D-fragment: v=0..7 -> consecutive p, fixed e => two float4s per tile.
    // Wave 0 initializes (covers all 2048 slots), waves 1..7 accumulate.
    if (wave == 0) {
      #pragma unroll
      for (int pt = 0; pt < 4; ++pt)
        #pragma unroll
        for (int et = 0; et < 2; ++et) {
          const int base = (et * 16 + ln16) * NP + pt * 16 + hi8;
          v4f lo, hh;
          #pragma unroll
          for (int v = 0; v < 4; ++v) { lo[v] = dAcc[pt][et][v];
                                        hh[v] = dAcc[pt][et][v + 4]; }
          *(v4f*)&s_delta[base]     = lo;
          *(v4f*)&s_delta[base + 4] = hh;
        }
    }
    __syncthreads();
    #pragma unroll 1
    for (int w = 1; w < NWAVES; ++w) {
      if (wave == w) {
        #pragma unroll
        for (int pt = 0; pt < 4; ++pt)
          #pragma unroll
          for (int et = 0; et < 2; ++et) {
            const int base = (et * 16 + ln16) * NP + pt * 16 + hi8;
            v4f lo = *(const v4f*)&s_delta[base];
            v4f hh = *(const v4f*)&s_delta[base + 4];
            #pragma unroll
            for (int v = 0; v < 4; ++v) { lo[v] += dAcc[pt][et][v];
                                          hh[v] += dAcc[pt][et][v + 4]; }
            *(v4f*)&s_delta[base]     = lo;
            *(v4f*)&s_delta[base + 4] = hh;
          }
      }
      __syncthreads();
    }

    // ---- prev <- delta ; full_state += ((T-t)/T) * delta ----
    const float wt = (float)(NT - t) / (float)NT;
    #pragma unroll
    for (int j = 0; j < 8; ++j) {
      int i = tid * 8 + j;                       // i = p*NE + e
      float d = s_delta[(i & 31) * NP + (i >> 5)];  // transposed read
      fullacc[j] += wt * d;
      s_prev[i] = (f16)d;
    }
    __syncthreads();
  }  // t

  // ---- stage full_state (natural (p,e) order), then out/base GEMVs --------
  #pragma unroll
  for (int j = 0; j < 8; ++j) s_delta[tid * 8 + j] = fullacc[j];
  __syncthreads();

  const int tg = tid;  // 256 threads == 256 targets
  float accO = b_out[tg];
  float accB = accO;
  for (int i = 0; i < NP * NE; ++i) {
    float w = W_out[(size_t)i * NTGT + tg];  // coalesced across threads
    accO += s_delta[i] * w;                  // full_state row (LDS broadcast)
    accB += mean_state[i] * w;               // ms row (uniform -> SMEM)
  }
  out[b * NTGT + tg] = accO;                 // out  (B,1,256,1)
  out[NB * NTGT + b * NTGT + tg] = accB;     // base (B,1,256,1)
}

extern "C" void kernel_launch(void* const* d_in, const int* in_sizes, int n_in,
                              void* d_out, int out_size, void* d_ws,
                              size_t ws_size, hipStream_t stream) {
  (void)in_sizes; (void)n_in; (void)out_size; (void)d_ws; (void)ws_size;
  const float* obs        = (const float*)d_in[0];
  // d_in[1] = target_positions : unused by the reference computation
  const float* mean_state = (const float*)d_in[2];
  const float* W_obs      = (const float*)d_in[3];
  const float* b_obs      = (const float*)d_in[4];
  const float* W_out      = (const float*)d_in[5];
  const float* b_out      = (const float*)d_in[6];
  suster_fwd<<<NB, THREADS, 0, stream>>>(obs, mean_state, W_obs, b_obs, W_out,
                                         b_out, (float*)d_out);
}
